// Architecture_47553877901676
// MI455X (gfx1250) — compile-verified, matrix operand order checked by reading, not run on hardware
//
#include <hip/hip_runtime.h>
#include <hip/hip_bf16.h>
#include <math.h>

typedef __attribute__((ext_vector_type(16))) _Float16 v16h;
typedef __attribute__((ext_vector_type(8)))  float    v8f;

#define BB   32
#define SS   256
#define DD   256
#define HH   8
#define DFFF 1024
#define NBL  2
#define MROWS (BB*SS)   // 8192

#if __has_builtin(__builtin_amdgcn_global_load_async_to_lds_b32)
#define HAVE_ASYNC_LDS 1
typedef __attribute__((address_space(1))) int gas_int;   // global
typedef __attribute__((address_space(3))) int las_int;   // LDS
#else
#define HAVE_ASYNC_LDS 0
#endif

#if __has_builtin(__builtin_amdgcn_tensor_load_to_lds)
#define HAVE_TDM 1
typedef unsigned int u32x4 __attribute__((ext_vector_type(4)));
typedef int          i32x4 __attribute__((ext_vector_type(4)));
typedef int          i32x8 __attribute__((ext_vector_type(8)));
#else
#define HAVE_TDM 0
#endif

static __device__ inline void wait_async_lds()
{
#if HAVE_ASYNC_LDS
#if __has_builtin(__builtin_amdgcn_s_wait_asynccnt)
    __builtin_amdgcn_s_wait_asynccnt(0);
#else
    asm volatile("s_wait_asynccnt 0" ::: "memory");
#endif
#endif
}

// ======================================================================
// Tiled WMMA GEMM:  C[M,N] = act( A[M,K] @ B[K,N] + bias + resid )
// Block tile 128x128 (BK=32), 8 waves, wave tile 32x64 = 2x4 WMMA tiles
// -> 8 x v_wmma_f32_16x16x32_f16 per wave per K step.
// K must be a multiple of 32 (true for every GEMM in this model).
// act: 0 = none, 1 = relu, 2 = sigmoid
// ======================================================================
__global__ __launch_bounds__(256)
void gemm_f16_wmma(const float* __restrict__ A, const float* __restrict__ B,
                   const float* __restrict__ bias, const float* __restrict__ resid,
                   float* __restrict__ C, int M, int N, int K, int act)
{
    // 40-half row pitch: each aligned 8-half group is 16B aligned -> ds_b128
    __shared__ _Float16 Ah[128][40];   // row-major A tile  [m][k]
    __shared__ _Float16 Bh[128][40];   // transposed B tile [n][k]

    const int tid  = threadIdx.x;
    const int lane = tid & 31;
    const int wv   = tid >> 5;         // 0..7
    const int wm   = (wv & 3) * 32;    // wave M offset within block tile
    const int wn   = (wv >> 2) * 64;   // wave N offset within block tile
    const int bm   = blockIdx.y * 128;
    const int bn   = blockIdx.x * 128;

    const int lh  = lane & 15;         // row (A) / col (B) within 16
    const int khi = (lane >> 4) * 8;   // K sub-offset: 0 (lanes 0-15) or 8

    const bool fullM = (bm + 128 <= M);
    const bool fullN = (bn + 128 <= N);

    v8f acc[2][4];
    #pragma unroll
    for (int im = 0; im < 2; ++im)
        #pragma unroll
        for (int in = 0; in < 4; ++in)
            #pragma unroll
            for (int e = 0; e < 8; ++e) acc[im][in][e] = 0.f;

    for (int kk = 0; kk < K; kk += 32) {
        // ---- prefetch next K tile (global_prefetch_b8) ----
        if (kk + 32 < K) {
            __builtin_prefetch(A + (size_t)(bm + (tid >> 1)) * K + (kk + 32) + (tid & 1) * 16, 0, 1);
            __builtin_prefetch(B + (size_t)(kk + 32 + (tid >> 5) * 4) * N + bn + (tid & 31) * 4, 0, 1);
        }

        // ---- stage A tile (128x32) ----
        if (fullM) {
            #pragma unroll
            for (int i = 0; i < 4; ++i) {
                int g  = tid + i * 256;          // float4 group 0..1023
                int r  = g >> 3;                 // 0..127
                int c4 = (g & 7) * 4;            // 0..28 step 4
                const float4 v = *(const float4*)(A + (size_t)(bm + r) * K + kk + c4);
                _Float16* dst = &Ah[r][c4];
                dst[0] = (_Float16)v.x; dst[1] = (_Float16)v.y;
                dst[2] = (_Float16)v.z; dst[3] = (_Float16)v.w;
            }
        } else {
            // clamp addresses; select value -> unconditional batched loads
            #pragma unroll
            for (int i = 0; i < 16; ++i) {
                int idx = tid + i * 256;
                int r = idx >> 5, c = idx & 31;
                int gm = bm + r;
                int cm = gm < M ? gm : (M - 1);
                float v = A[(size_t)cm * K + kk + c];
                Ah[r][c] = (_Float16)(gm < M ? v : 0.f);
            }
        }
        // ---- stage B tile (32x128), transposed into LDS ----
        if (fullN) {
            #pragma unroll
            for (int i = 0; i < 4; ++i) {
                int g  = tid + i * 256;          // group: 4 K x 1 N
                int n  = g & 127;
                int k4 = (g >> 7) * 4;           // 0..28 step 4
                const float* src = B + (size_t)(kk + k4) * N + bn + n;
                float x0 = src[0];
                float x1 = src[(size_t)N];
                float x2 = src[2 * (size_t)N];
                float x3 = src[3 * (size_t)N];
                _Float16* dst = &Bh[n][k4];
                dst[0] = (_Float16)x0; dst[1] = (_Float16)x1;
                dst[2] = (_Float16)x2; dst[3] = (_Float16)x3;
            }
        } else {
            #pragma unroll
            for (int i = 0; i < 16; ++i) {
                int idx = tid + i * 256;
                int k = idx >> 7, n = idx & 127;
                int gn = bn + n;
                int cn = gn < N ? gn : (N - 1);
                float v = B[(size_t)(kk + k) * N + cn];
                Bh[n][k] = (_Float16)(gn < N ? v : 0.f);
            }
        }
        __syncthreads();

        // ---- fragments per ISA 16-bit layout (K 0-7/16-23 | 8-15/24-31) ----
        v16h af[2], bf[4];
        #pragma unroll
        for (int im = 0; im < 2; ++im) {
            int r = wm + im * 16 + lh;
            #pragma unroll
            for (int e = 0; e < 8; ++e) {
                af[im][e]     = Ah[r][khi + e];
                af[im][e + 8] = Ah[r][16 + khi + e];
            }
        }
        #pragma unroll
        for (int in = 0; in < 4; ++in) {
            int cn = wn + in * 16 + lh;
            #pragma unroll
            for (int e = 0; e < 8; ++e) {
                bf[in][e]     = Bh[cn][khi + e];
                bf[in][e + 8] = Bh[cn][16 + khi + e];
            }
        }
        #pragma unroll
        for (int im = 0; im < 2; ++im)
            #pragma unroll
            for (int in = 0; in < 4; ++in)
                acc[im][in] = __builtin_amdgcn_wmma_f32_16x16x32_f16(
                    false, af[im], false, bf[in], (short)0, acc[im][in],
                    false, false);
        __syncthreads();
    }

    // ---- epilogue: VGPR e -> M=e (lanes 0-15) / M=e+8 (lanes 16-31) ----
    const int rbase = (lane >> 4) * 8;
    #pragma unroll
    for (int im = 0; im < 2; ++im) {
        #pragma unroll
        for (int in = 0; in < 4; ++in) {
            int col = bn + wn + in * 16 + lh;
            if (!fullN && col >= N) continue;
            float bv = bias ? bias[col] : 0.f;
            #pragma unroll
            for (int e = 0; e < 8; ++e) {
                int row = bm + wm + im * 16 + rbase + e;
                if (!fullM && row >= M) continue;
                float v = acc[im][in][e] + bv;
                if (resid) v += resid[(size_t)row * N + col];
                if (act == 1)      v = v > 0.f ? v : 0.f;
                else if (act == 2) v = 1.f / (1.f + __expf(-v));
                C[(size_t)row * N + col] = v;
            }
        }
    }
}

// ======================================================================
// Attention: online softmax per query row, dif_mask applied post-softmax
// (outside the normalizer, per reference). One block per (b,h); thread = q.
// ======================================================================
#define ATT_HD 32
__global__ __launch_bounds__(256)
void attn_kernel(const float* __restrict__ qkv, const float* __restrict__ dif,
                 float* __restrict__ o)
{
    __shared__ _Float16 Ks[SS][ATT_HD + 2];
    __shared__ _Float16 Vs[SS][ATT_HD + 2];
    const int b = blockIdx.x / HH;
    const int h = blockIdx.x % HH;
    const int t = threadIdx.x;          // query row (S == blockDim.x == 256)
    const int ld = 3 * DD;

    const size_t rowbase = ((size_t)b * SS + t) * ld;
    #pragma unroll
    for (int d = 0; d < ATT_HD; ++d) {
        Ks[t][d] = (_Float16)qkv[rowbase + DD     + h * ATT_HD + d];
        Vs[t][d] = (_Float16)qkv[rowbase + 2 * DD + h * ATT_HD + d];
    }
    float qv[ATT_HD];
    #pragma unroll
    for (int d = 0; d < ATT_HD; ++d) qv[d] = qkv[rowbase + h * ATT_HD + d];
    __syncthreads();

    float m = -1e30f, l = 0.f;
    float accv[ATT_HD];
    #pragma unroll
    for (int d = 0; d < ATT_HD; ++d) accv[d] = 0.f;

    const float scale = 0.17677669529663689f;  // 1/sqrt(32)
    const float* difrow = dif + ((size_t)b * SS + t) * SS;

    for (int k = 0; k <= t; ++k) {              // causal
        float s = 0.f;
        #pragma unroll
        for (int d = 0; d < ATT_HD; ++d) s += qv[d] * (float)Ks[k][d];
        s *= scale;
        float nm = fmaxf(m, s);
        float sc = __expf(m - nm);
        float e  = __expf(s - nm);
        l = l * sc + e;
        float w = e * difrow[k];
        #pragma unroll
        for (int d = 0; d < ATT_HD; ++d) accv[d] = accv[d] * sc + w * (float)Vs[k][d];
        m = nm;
    }
    const float inv = 1.f / l;
    const size_t ob = ((size_t)b * SS + t) * DD + h * ATT_HD;
    #pragma unroll
    for (int d = 0; d < ATT_HD; ++d) o[ob + d] = accv[d] * inv;
}

// ======================================================================
// LayerNorm over width 256 (one block per row); row staged into the LDS
// reduction buffer via gfx1250 async-LDS load.
// ======================================================================
__global__ __launch_bounds__(256)
void ln_kernel(const float* __restrict__ x, const float* __restrict__ w,
               const float* __restrict__ bv, float* __restrict__ out)
{
    const int row = blockIdx.x, t = threadIdx.x;
    __shared__ float red[256];
#if HAVE_ASYNC_LDS
    __builtin_amdgcn_global_load_async_to_lds_b32(
        (gas_int*)(x + (size_t)row * DD + t), (las_int*)&red[t], 0, 0);
    wait_async_lds();
    __syncthreads();
    float v = red[t];
#else
    float v = x[(size_t)row * DD + t];
    red[t] = v; __syncthreads();
#endif
    for (int o = 128; o > 0; o >>= 1) { if (t < o) red[t] += red[t + o]; __syncthreads(); }
    float mean = red[0] * (1.f / 256.f);
    __syncthreads();
    float d = v - mean;
    red[t] = d * d; __syncthreads();
    for (int o = 128; o > 0; o >>= 1) { if (t < o) red[t] += red[t + o]; __syncthreads(); }
    float var = red[0] * (1.f / 256.f);
    out[(size_t)row * DD + t] = d * rsqrtf(var + 1e-5f) * w[t] + bv[t];
}

// ======================================================================
// dt = softplus(zxbcdt[..., -nh:] + dt_bias)
// ======================================================================
__global__ void dt_kernel(const float* __restrict__ zx, const float* __restrict__ dtb,
                          float* __restrict__ dt, int ldp, int nh, int total)
{
    int i = blockIdx.x * blockDim.x + threadIdx.x;
    if (i >= total) return;
    int row = i / nh, h = i % nh;
    float x = zx[(size_t)row * ldp + (ldp - nh) + h] + dtb[h];
    dt[i] = (x > 20.f) ? x : log1pf(__expf(x));
}

// ======================================================================
// Depthwise causal conv (DCONV=4) + SiLU over xBC channels
// ======================================================================
__global__ void conv_silu_kernel(const float* __restrict__ zx, const float* __restrict__ cw,
                                 const float* __restrict__ cb, float* __restrict__ xbc,
                                 int ldp, int d_inner, int cd, int total)
{
    int i = blockIdx.x * blockDim.x + threadIdx.x;
    if (i >= total) return;
    int c = i % cd;
    int r = i / cd;
    int s = r % SS, b = r / SS;
    float acc = cb[c];
    #pragma unroll
    for (int k = 0; k < 4; ++k) {
        int sp = s + k - 3;
        if (sp >= 0)
            acc += zx[(size_t)(b * SS + sp) * ldp + d_inner + c] * cw[c * 4 + k];
    }
    xbc[(size_t)r * cd + c] = acc / (1.f + __expf(-acc));
}

// ======================================================================
// SSM scan: one block per (b, head). HEADDIM=64, DSTATE=128.
// thread t: p = t>>2, owns 32 of 128 state cols; quad-reduce for y.
// B/C state staged in CHUNK-step tiles: one TDM tensor_load_to_lds per
// chunk (2D tile 256 x CHUNK fp32, row stride cd) -> zero per-step barriers.
// ======================================================================
#define SCHUNK 16
__global__ __launch_bounds__(256)
void scan_kernel(const float* __restrict__ xbc, const float* __restrict__ dt,
                 const float* __restrict__ A_log, const float* __restrict__ Dp,
                 float* __restrict__ y, int nh, int cd, int d_inner)
{
    const int head = blockIdx.x % nh;
    const int b    = blockIdx.x / nh;
    const int t    = threadIdx.x;
    const int p    = t >> 2;
    const int n0   = (t & 3) * 32;
    const float Aneg = -__expf(A_log[head]);
    const float Dh   = Dp[head];

    float hst[32];
    #pragma unroll
    for (int j = 0; j < 32; ++j) hst[j] = 0.f;

    __shared__ float sTile[SCHUNK * 256];   // per step: [0..127]=B_t, [128..255]=C_t

    for (int s0 = 0; s0 < SS; s0 += SCHUNK) {
        __syncthreads();   // previous chunk fully consumed
#if HAVE_TDM
        if (t < 32) {      // wave 0 issues the tensor DMA
            const size_t gaddr = (size_t)(const void*)(xbc + ((size_t)b * SS + s0) * cd + d_inner);
            const unsigned lds = (unsigned)(size_t)(void*)&sTile[0];
            // D# group0: count=1 | lds_addr | global_addr[56:0] | type=2
            u32x4 g0 = { 1u, lds, (unsigned)(gaddr & 0xFFFFFFFFu),
                         (unsigned)((gaddr >> 32) & 0x01FFFFFFu) | 0x80000000u };
            // D# group1: data_size=4B; tensor 256 x SCHUNK; tile 256 x SCHUNK; dim0 stride = cd
            i32x8 g1 = { (int)0x00020000u,                       // data_size=2 (4 bytes)
                         (int)(256u << 16),                      // tensor_dim0[15:0]=256
                         (int)((unsigned)SCHUNK << 16),          // tensor_dim1[15:0]=SCHUNK
                         (int)(256u << 16),                      // tile_dim0=256
                         (int)SCHUNK,                            // tile_dim1=SCHUNK, tile_dim2=0
                         (int)cd,                                // tensor_dim0_stride[31:0]
                         0, 0 };
            i32x4 g2 = { 0, 0, 0, 0 };
            i32x4 g3 = { 0, 0, 0, 0 };
            i32x8 g4 = { 0, 0, 0, 0, 0, 0, 0, 0 };
            __builtin_amdgcn_tensor_load_to_lds(g0, g1, g2, g3, g4, 0);
            __builtin_amdgcn_s_wait_tensorcnt(0);
        }
        __syncthreads();
#else
        #pragma unroll
        for (int j = 0; j < SCHUNK; ++j) {
            const float* gsrc = xbc + ((size_t)b * SS + s0 + j) * cd + d_inner + t;
#if HAVE_ASYNC_LDS
            __builtin_amdgcn_global_load_async_to_lds_b32(
                (gas_int*)gsrc, (las_int*)&sTile[j * 256 + t], 0, 0);
#else
            sTile[j * 256 + t] = *gsrc;
#endif
        }
        wait_async_lds();
        __syncthreads();
#endif
        for (int cs = 0; cs < SCHUNK; ++cs) {
            const int s = s0 + cs;
            const float* sB = &sTile[cs * 256];
            const size_t base = ((size_t)b * SS + s) * cd;
            const float dtv = dt[((size_t)b * SS + s) * nh + head];
            const float a   = __expf(dtv * Aneg);
            const float xv  = xbc[base + head * 64 + p];
            const float xdt = xv * dtv;

            float part = 0.f;
            #pragma unroll
            for (int j = 0; j < 32; ++j) {
                hst[j] = a * hst[j] + xdt * sB[n0 + j];
                part  += hst[j] * sB[128 + n0 + j];
            }
            part += __shfl_xor(part, 1, 32);
            part += __shfl_xor(part, 2, 32);
            if ((t & 3) == 0)
                y[((size_t)b * SS + s) * d_inner + head * 64 + p] = part + xv * Dh;
        }
    }
}

// ======================================================================
// y = RMSNorm(y * silu(z)) * norm_w   (in place on y); W = d_inner
// ======================================================================
__global__ __launch_bounds__(256)
void gate_rms_kernel(float* __restrict__ y, const float* __restrict__ zx,
                     const float* __restrict__ nw, int ldp, int W)
{
    const int row = blockIdx.x, t = threadIdx.x;
    const int E = W / 256;
    float g[8];
    float ss = 0.f;
    for (int e = 0; e < E; ++e) {
        int c = t + e * 256;
        float z  = zx[(size_t)row * ldp + c];
        float sz = z / (1.f + __expf(-z));
        float v  = y[(size_t)row * W + c] * sz;
        g[e] = v; ss += v * v;
    }
    __shared__ float red[256];
    red[t] = ss; __syncthreads();
    for (int o = 128; o > 0; o >>= 1) { if (t < o) red[t] += red[t + o]; __syncthreads(); }
    float sc = rsqrtf(red[0] / (float)W + 1e-5f);
    for (int e = 0; e < E; ++e) {
        int c = t + e * 256;
        y[(size_t)row * W + c] = g[e] * sc * nw[c];
    }
}

// ======================================================================
// xa = concat([qa, kca, shift(q), shift(kc)], -1)  (width 4*D)
// ======================================================================
__global__ void concat_kernel(const float* __restrict__ qa, const float* __restrict__ kca,
                              const float* __restrict__ q, const float* __restrict__ kc,
                              float* __restrict__ xa, int total)
{
    int i = blockIdx.x * blockDim.x + threadIdx.x;
    if (i >= total) return;
    int col = i % (4 * DD);
    int r   = i / (4 * DD);
    int s   = r % SS;
    float v;
    if (col < DD)               v = qa[(size_t)r * DD + col];
    else if (col < 2 * DD)      v = kca[(size_t)r * DD + (col - DD)];
    else if (col < 3 * DD)      v = (s + 1 < SS) ? q[(size_t)(r + 1) * DD + (col - 2 * DD)] : 0.f;
    else                        v = (s + 1 < SS) ? kc[(size_t)(r + 1) * DD + (col - 3 * DD)] : 0.f;
    xa[i] = v;
}

// ======================================================================
// Host-side orchestration
// ======================================================================
static inline void gemm(hipStream_t st, const float* A, const float* B,
                        const float* bias, const float* resid, float* C,
                        int M, int N, int K, int act)
{
    dim3 grid((N + 127) / 128, (M + 127) / 128);
    gemm_f16_wmma<<<grid, 256, 0, st>>>(A, B, bias, resid, C, M, N, K, act);
}

static void run_mamba(hipStream_t st, const float* u, int dm, const float* const* p,
                      float* zx, float* dtbuf, float* xbc, float* ybuf, float* out)
{
    const int d_inner = 2 * dm;
    const int nh      = d_inner / 64;
    const int cd      = d_inner + 256;               // d_inner + 2*DSTATE
    const int ldp     = 2 * d_inner + 256 + nh;      // d_in_proj
    gemm(st, u, p[0], nullptr, nullptr, zx, MROWS, ldp, dm, 0);
    int tdt = MROWS * nh;
    dt_kernel<<<(tdt + 255) / 256, 256, 0, st>>>(zx, p[3], dtbuf, ldp, nh, tdt);
    int tc = MROWS * cd;
    conv_silu_kernel<<<(tc + 255) / 256, 256, 0, st>>>(zx, p[1], p[2], xbc, ldp, d_inner, cd, tc);
    scan_kernel<<<BB * nh, 256, 0, st>>>(xbc, dtbuf, p[4], p[5], ybuf, nh, cd, d_inner);
    gate_rms_kernel<<<MROWS, 256, 0, st>>>(ybuf, zx, p[6], ldp, d_inner);
    gemm(st, ybuf, p[7], nullptr, nullptr, out, MROWS, dm, d_inner, 0);
}

extern "C" void kernel_launch(void* const* d_in, const int* in_sizes, int n_in,
                              void* d_out, int out_size, void* d_ws, size_t ws_size,
                              hipStream_t stream)
{
    (void)in_sizes; (void)n_in; (void)out_size; (void)ws_size;

    // ---- workspace layout (MB offsets) ----
    char* ws = (char*)d_ws;
    auto wsf = [&](size_t mb) { return (float*)(ws + mb * (size_t)(1 << 20)); };
    float* Sout[4] = { wsf(0), wsf(8), wsf(16), wsf(24) };   // stack outputs
    float* xa    = wsf(32);
    // stage-1 temporaries (region reused by stage 2 afterwards)
    float* qkv   = wsf(64);
    float* attno = wsf(88);
    float* y1    = wsf(96);
    float* x1    = wsf(104);
    float* f1    = wsf(112);
    float* y2    = wsf(144);
    // stage-2 buffers
    float* zx5   = wsf(64);
    float* dt5   = wsf(208);
    float* xbc5  = wsf(209);
    float* y5    = wsf(286);
    float* m5out = wsf(350);
    float* hbuf  = wsf(382);
    float* zx6   = wsf(390);
    float* dt6   = wsf(433);
    float* xbc6  = wsf(434);
    float* y6    = wsf(458);

    // ---- inputs (flattened depth-first in setup_inputs() insertion order) ----
    const float* embeds[4] = { (const float*)d_in[0], (const float*)d_in[2],
                               (const float*)d_in[1], (const float*)d_in[3] };
    const float* masks[4]  = { (const float*)d_in[4], (const float*)d_in[5],
                               (const float*)d_in[4], (const float*)d_in[5] };

    // ---- 4 transformer stacks ----
    for (int s = 0; s < 4; ++s) {
        const int pb = 6 + s * 11;
        const float* Wqkv = (const float*)d_in[pb + 0];
        const float* Wo   = (const float*)d_in[pb + 1];
        const float* bo   = (const float*)d_in[pb + 2];
        const float* l1w  = (const float*)d_in[pb + 3];
        const float* l1b  = (const float*)d_in[pb + 4];
        const float* W1   = (const float*)d_in[pb + 5];
        const float* b1   = (const float*)d_in[pb + 6];
        const float* W2   = (const float*)d_in[pb + 7];
        const float* b2   = (const float*)d_in[pb + 8];
        const float* l2w  = (const float*)d_in[pb + 9];
        const float* l2b  = (const float*)d_in[pb + 10];

        const float* x = embeds[s];
        float* xout = Sout[s];
        for (int i = 0; i < NBL; ++i) {
            gemm(stream, x, Wqkv + (size_t)i * DD * 3 * DD, nullptr, nullptr,
                 qkv, MROWS, 3 * DD, DD, 0);
            attn_kernel<<<BB * HH, 256, 0, stream>>>(qkv, masks[s], attno);
            gemm(stream, attno, Wo + (size_t)i * DD * DD, bo + (size_t)i * DD,
                 x, y1, MROWS, DD, DD, 0);
            ln_kernel<<<MROWS, 256, 0, stream>>>(y1, l1w + i * DD, l1b + i * DD, x1);
            gemm(stream, x1, W1 + (size_t)i * DD * DFFF, b1 + (size_t)i * DFFF,
                 nullptr, f1, MROWS, DFFF, DD, 1);
            gemm(stream, f1, W2 + (size_t)i * DFFF * DD, b2 + (size_t)i * DD,
                 x1, y2, MROWS, DD, DFFF, 0);
            ln_kernel<<<MROWS, 256, 0, stream>>>(y2, l2w + i * DD, l2b + i * DD, xout);
            x = xout;
        }
    }

    // ---- xa = concat([qa, kca, shift(q), shift(kc)]) ----
    {
        int tcat = MROWS * 4 * DD;
        concat_kernel<<<(tcat + 255) / 256, 256, 0, stream>>>(
            Sout[1], Sout[3], Sout[0], Sout[2], xa, tcat);
    }

    // ---- mamba5 (d_model = 1024) ----
    const float* m5p[8];
    for (int j = 0; j < 8; ++j) m5p[j] = (const float*)d_in[50 + j];
    run_mamba(stream, xa, 4 * DD, m5p, zx5, dt5, xbc5, y5, m5out);

    // ---- h = sigmoid(m5out @ lin_W + lin_b) ----
    gemm(stream, m5out, (const float*)d_in[66], (const float*)d_in[67],
         nullptr, hbuf, MROWS, DD, 4 * DD, 2);

    // ---- mamba6 (d_model = 256) -> d_out ----
    const float* m6p[8];
    for (int j = 0; j < 8; ++j) m6p[j] = (const float*)d_in[58 + j];
    run_mamba(stream, hbuf, DD, m6p, zx6, dt6, xbc6, y6, (float*)d_out);
}
